// SileroVAD_83829171683562
// MI455X (gfx1250) — compile-verified, matrix operand order checked by reading, not run on hardware
//
#include <hip/hip_runtime.h>
#include <hip/hip_bf16.h>
#include <math.h>

typedef _Float16 v16h __attribute__((ext_vector_type(16)));
typedef _Float16 h8   __attribute__((ext_vector_type(8)));
typedef float    v8f  __attribute__((ext_vector_type(8)));

// ---------------- packed weight layout (offsets in _Float16 units) -------------
// Fragment = 512 halfs (32 lanes x 16 halfs), laid out so lane L reads 32
// contiguous bytes at frag*1024 + L*32.
#define OFF_STFT_H  0u        // 17 ntiles x 8 ksteps   = 136 frags
#define OFF_E1_H    69632u    // 3 taps x 8 nt x 5 ks   = 120 frags
#define OFF_E2_H    131072u   // 3 taps x 4 nt x 4 ks   =  48 frags
#define OFF_E3_H    155648u   // 2 taps x 4 nt x 2 ks   =  16 frags
#define OFF_E4_H    163840u   // 8 nt x 2 ks            =  16 frags
#define OFF_LSTM_H  172032u   // 32 nt x 4 ks           = 128 frags
#define TOTAL_H     237568u   // 475136 bytes in d_ws

// ---------------- LDS arena (bytes) -------------------------------------------
#define RS_X     648   // halfs per row of x   [16][648]
#define RS_SPEC  280   // f32  per row of spec [16][280]  (reused as h5[16][128])
#define RS_MAG   168   // halfs per row        [6][16][168]
#define RS_H1    136   //                      [6][16][136]
#define RS_H2    72    //                      [4][16][72]
#define RS_H3    72    //                      [16][72]
#define RS_H4    136   //                      [16][136]
#define RS_G     520   // f32                  [16][520]

#define OFF_X     0u
#define OFF_SPEC  20736u
#define OFF_MAG   38656u
#define OFF_H1    70912u   // mag,h1,h2 contiguous -> one zero-fill loop
#define OFF_H2    97024u
#define OFF_H3    106240u
#define OFF_H4    108544u
#define OFF_G     112896u
#define SMEM_BYTES 146176u

// =============================================================================
// Prepack: W(N x K) -> WMMA B fragments (f16), zero-padded outside N/K.
// element = src[n*rowStride + k*kStride + base0 + dt*dtStride]
// frag id b = ((dt*ntiles + ntile)*ksteps + s)
// B layout per 16x16x32 wmma: lane L holds col n = L&15,
//   k = 32*s + 8*(L>>4) + (h&7) + 16*(h>>3)   for half h = 0..15
// =============================================================================
__global__ void pack_b_frag(const float* __restrict__ src, _Float16* __restrict__ dst,
                            int ntiles, int ksteps, int N, int K,
                            int rowStride, int kStride, int base0, int dtStride)
{
    int b  = blockIdx.x;
    int s  = b % ksteps;
    int t1 = b / ksteps;
    int nt = t1 % ntiles;
    int dt = t1 / ntiles;
    int L  = threadIdx.x;

    int n  = nt * 16 + (L & 15);
    int kb = 32 * s + 8 * (L >> 4);
    _Float16* out = dst + (size_t)b * 512 + (size_t)L * 16;
#pragma unroll
    for (int h = 0; h < 16; ++h) {
        int k = kb + (h & 7) + 16 * (h >> 3);
        float v = 0.f;
        if (n < N && k < K)
            v = src[(size_t)n * rowStride + (size_t)k * kStride + base0 + dt * dtStride];
        out[h] = (_Float16)v;
    }
}

// =============================================================================
// Fused VAD kernel: 1 block = 16 batch rows, 4 waves cooperate via LDS.
// =============================================================================
__device__ __forceinline__ v8f wmma_f16(v16h a, v16h b, v8f c)
{
    return __builtin_amdgcn_wmma_f32_16x16x32_f16(false, a, false, b,
                                                  (short)0, c, false, false);
}

// A fragment (16x32 f16) from LDS rows of stride rs (halfs), k-base col.
__device__ __forceinline__ v16h ldsA(const _Float16* base, int rs, int lane, int col)
{
    const _Float16* p = base + (lane & 15) * rs + col + 8 * (lane >> 4);
    h8 lo = *(const h8*)p;        // K = col+8*hi + 0..7
    h8 hi = *(const h8*)(p + 16); // K = col+8*hi + 16..23
    v16h a;
#pragma unroll
    for (int h = 0; h < 8; ++h) { a[h] = lo[h]; a[h + 8] = hi[h]; }
    return a;
}

__device__ __forceinline__ v16h ldB(const _Float16* wp, int lane)
{
    return *(const v16h*)(wp + (size_t)lane * 16); // 32B contiguous per lane
}

__global__ __launch_bounds__(128)
void vad_fused(const float* __restrict__ data,
               const _Float16* __restrict__ wpack,
               const float* __restrict__ e1_b, const float* __restrict__ e2_b,
               const float* __restrict__ e3_b, const float* __restrict__ e4_b,
               const float* __restrict__ b_ih, const float* __restrict__ b_hh,
               const float* __restrict__ dec_w, const float* __restrict__ dec_b,
               float* __restrict__ out)
{
    extern __shared__ char smem[];
    const int tid  = threadIdx.x;
    const int lane = tid & 31;
    const int wid  = tid >> 5;
    const int b0   = blockIdx.x * 16;
    const int nc   = lane & 15;
    const int rhi  = lane >> 4;

    _Float16* lx   = (_Float16*)(smem + OFF_X);
    float*    lsp  = (float*)(smem + OFF_SPEC);
    _Float16* lmag = (_Float16*)(smem + OFF_MAG);
    _Float16* lh1  = (_Float16*)(smem + OFF_H1);
    _Float16* lh2  = (_Float16*)(smem + OFF_H2);
    _Float16* lh3  = (_Float16*)(smem + OFF_H3);
    _Float16* lh4  = (_Float16*)(smem + OFF_H4);
    float*    lg   = (float*)(smem + OFF_G);

    // ---- phase 0: zero pad slots (mag/h1/h2 contiguous), stage x into LDS ----
    {
        uint32_t* z = (uint32_t*)lmag;
        const int nd = (6*16*RS_MAG + 6*16*RS_H1 + 4*16*RS_H2) / 2; // halfs->dwords
        for (int i = tid; i < nd; i += 128) z[i] = 0u;
    }
    for (int i = tid; i < 16*64; i += 128) {        // 64 zero-context samples
        int m = i >> 6, c = i & 63;
        lx[m * RS_X + c] = (_Float16)0.f;
    }
    for (int i = tid; i < 16*512; i += 128) {       // payload
        int m = i >> 9, c = i & 511;
        lx[m * RS_X + 64 + c] = (_Float16)data[(size_t)(b0 + m) * 512 + c];
    }
    for (int i = tid; i < 16*64; i += 128) {        // reflect tail: x[576+r]=d[510-r]
        int m = i >> 6, r = i & 63;
        lx[m * RS_X + 576 + r] = (_Float16)data[(size_t)(b0 + m) * 512 + 510 - r];
    }
    __syncthreads();

    // ---- phase 1: STFT conv (stride 128, K=256) + magnitude, per time step ---
    for (int t = 0; t < 4; ++t) {
        for (int nt = wid; nt < 17; nt += 4) {
            const _Float16* wp = wpack + OFF_STFT_H + (size_t)(nt * 8) * 512;
            __builtin_prefetch((const void*)wp, 0, 1);
            v8f c = {0.f,0.f,0.f,0.f,0.f,0.f,0.f,0.f};
#pragma unroll
            for (int s = 0; s < 8; ++s) {
                v16h a = ldsA(lx, RS_X, lane, 128 * t + 32 * s);
                v16h b = ldB(wp + (size_t)s * 512, lane);
                c = wmma_f16(a, b, c);
            }
#pragma unroll
            for (int r = 0; r < 8; ++r)
                lsp[(r + 8 * rhi) * RS_SPEC + nt * 16 + nc] = c[r];
        }
        __syncthreads();
        _Float16* mslot = lmag + (size_t)(t + 1) * 16 * RS_MAG;
        for (int n = tid; n < 129; n += 128) {
#pragma unroll
            for (int m = 0; m < 16; ++m) {
                float re = lsp[m * RS_SPEC + n];
                float im = lsp[m * RS_SPEC + 129 + n];
                mslot[m * RS_MAG + n] = (_Float16)sqrtf(re * re + im * im);
            }
        }
        __syncthreads();
    }

    // ---- phase 2: e1 conv (129->128, k=3, pad 1) : wave w owns t=w ------------
    {
        const int t = wid;
        for (int nt = 0; nt < 8; ++nt) {
            v8f c = {0.f,0.f,0.f,0.f,0.f,0.f,0.f,0.f};
            for (int dt = 0; dt < 3; ++dt) {
                const _Float16* ms = lmag + (size_t)(t + dt) * 16 * RS_MAG;
                const _Float16* wp = wpack + OFF_E1_H + (size_t)((dt * 8 + nt) * 5) * 512;
                __builtin_prefetch((const void*)wp, 0, 1);
#pragma unroll
                for (int s = 0; s < 5; ++s) {
                    v16h a = ldsA(ms, RS_MAG, lane, 32 * s);
                    v16h b = ldB(wp + (size_t)s * 512, lane);
                    c = wmma_f16(a, b, c);
                }
            }
            float bv = e1_b[nt * 16 + nc];
            _Float16* hs = lh1 + (size_t)(t + 1) * 16 * RS_H1;
#pragma unroll
            for (int r = 0; r < 8; ++r)
                hs[(r + 8 * rhi) * RS_H1 + nt * 16 + nc] =
                    (_Float16)fmaxf(c[r] + bv, 0.f);
        }
    }
    __syncthreads();

    // ---- phase 3: e2 conv (128->64, k=3, stride 2, pad 1), T 4 -> 2 ----------
    for (int q = wid; q < 8; q += 4) {
        int j = q >> 2, nt = q & 3;
        v8f c = {0.f,0.f,0.f,0.f,0.f,0.f,0.f,0.f};
        for (int dt = 0; dt < 3; ++dt) {
            const _Float16* hs = lh1 + (size_t)(2 * j + dt) * 16 * RS_H1;
            const _Float16* wp = wpack + OFF_E2_H + (size_t)((dt * 4 + nt) * 4) * 512;
#pragma unroll
            for (int s = 0; s < 4; ++s) {
                v16h a = ldsA(hs, RS_H1, lane, 32 * s);
                v16h b = ldB(wp + (size_t)s * 512, lane);
                c = wmma_f16(a, b, c);
            }
        }
        float bv = e2_b[nt * 16 + nc];
        _Float16* d2 = lh2 + (size_t)(j + 1) * 16 * RS_H2;
#pragma unroll
        for (int r = 0; r < 8; ++r)
            d2[(r + 8 * rhi) * RS_H2 + nt * 16 + nc] = (_Float16)fmaxf(c[r] + bv, 0.f);
    }
    __syncthreads();

    // ---- phase 4: e3 conv (64->64, k=3, stride 2, pad 1), T 2 -> 1 -----------
    {
        const int nt = wid; // 4 n-tiles, 4 waves
        v8f c = {0.f,0.f,0.f,0.f,0.f,0.f,0.f,0.f};
        for (int dtp = 0; dtp < 2; ++dtp) { // taps 1,2 (tap 0 hits zero pad)
            const _Float16* hs = lh2 + (size_t)(dtp + 1) * 16 * RS_H2;
            const _Float16* wp = wpack + OFF_E3_H + (size_t)((dtp * 4 + nt) * 2) * 512;
#pragma unroll
            for (int s = 0; s < 2; ++s) {
                v16h a = ldsA(hs, RS_H2, lane, 32 * s);
                v16h b = ldB(wp + (size_t)s * 512, lane);
                c = wmma_f16(a, b, c);
            }
        }
        float bv = e3_b[nt * 16 + nc];
#pragma unroll
        for (int r = 0; r < 8; ++r)
            lh3[(r + 8 * rhi) * RS_H3 + nt * 16 + nc] = (_Float16)fmaxf(c[r] + bv, 0.f);
    }
    __syncthreads();

    // ---- phase 5: e4 conv (64->128, k=3, pad 1, L=1 -> center tap only) ------
    for (int nt = wid; nt < 8; nt += 4) {
        const _Float16* wp = wpack + OFF_E4_H + (size_t)(nt * 2) * 512;
        v8f c = {0.f,0.f,0.f,0.f,0.f,0.f,0.f,0.f};
#pragma unroll
        for (int s = 0; s < 2; ++s) {
            v16h a = ldsA(lh3, RS_H3, lane, 32 * s);
            v16h b = ldB(wp + (size_t)s * 512, lane);
            c = wmma_f16(a, b, c);
        }
        float bv = e4_b[nt * 16 + nc];
#pragma unroll
        for (int r = 0; r < 8; ++r)
            lh4[(r + 8 * rhi) * RS_H4 + nt * 16 + nc] = (_Float16)fmaxf(c[r] + bv, 0.f);
    }
    __syncthreads();

    // ---- phase 6: LSTM gates: xt @ w_ih^T + (b_ih + b_hh)  (h0 = 0) ----------
    for (int nt = wid; nt < 32; nt += 4) {
        const _Float16* wp = wpack + OFF_LSTM_H + (size_t)(nt * 4) * 512;
        __builtin_prefetch((const void*)wp, 0, 1);
        v8f c = {0.f,0.f,0.f,0.f,0.f,0.f,0.f,0.f};
#pragma unroll
        for (int s = 0; s < 4; ++s) {
            v16h a = ldsA(lh4, RS_H4, lane, 32 * s);
            v16h b = ldB(wp + (size_t)s * 512, lane);
            c = wmma_f16(a, b, c);
        }
        int n0 = nt * 16;
        float bv = b_ih[n0 + nc] + b_hh[n0 + nc];
#pragma unroll
        for (int r = 0; r < 8; ++r)
            lg[(r + 8 * rhi) * RS_G + n0 + nc] = c[r] + bv;
    }
    __syncthreads();

    // ---- phase 7: LSTM nonlinearity + relu(h) (c0 = 0 kills the f-gate) ------
    float* h5 = lsp; // reuse spec region as [16][128] f32
    for (int i = tid; i < 2048; i += 128) {
        int m = i >> 7, n = i & 127;
        const float* g = lg + m * RS_G;
        float ig = g[n], gg = g[256 + n], og = g[384 + n];
        float cc = (1.f / (1.f + expf(-ig))) * tanhf(gg);
        float hh = (1.f / (1.f + expf(-og))) * tanhf(cc);
        h5[m * 128 + n] = fmaxf(hh, 0.f);
    }
    __syncthreads();

    // ---- phase 8: decoder: sigmoid(h . dec_w + dec_b) ------------------------
    if (tid < 16) {
        float acc = dec_b[0];
#pragma unroll 8
        for (int n = 0; n < 128; ++n) acc += h5[tid * 128 + n] * dec_w[n];
        out[b0 + tid] = 1.f / (1.f + expf(-acc));
    }
}

// =============================================================================
extern "C" void kernel_launch(void* const* d_in, const int* in_sizes, int n_in,
                              void* d_out, int out_size, void* d_ws, size_t ws_size,
                              hipStream_t stream)
{
    const float* data   = (const float*)d_in[0];
    // d_in[1] = sr (16000 -> context 64, baked in)
    const float* stft_w = (const float*)d_in[2];
    const float* e1_w   = (const float*)d_in[3];
    const float* e1_b   = (const float*)d_in[4];
    const float* e2_w   = (const float*)d_in[5];
    const float* e2_b   = (const float*)d_in[6];
    const float* e3_w   = (const float*)d_in[7];
    const float* e3_b   = (const float*)d_in[8];
    const float* e4_w   = (const float*)d_in[9];
    const float* e4_b   = (const float*)d_in[10];
    const float* w_ih   = (const float*)d_in[11];
    // d_in[12] = w_hh: unused (h0 == 0)
    const float* b_ih   = (const float*)d_in[13];
    const float* b_hh   = (const float*)d_in[14];
    const float* dec_w  = (const float*)d_in[15];
    const float* dec_b  = (const float*)d_in[16];
    float* out = (float*)d_out;
    _Float16* wp = (_Float16*)d_ws;

    // -------- pack all weights into f16 WMMA B-fragments in d_ws -------------
    //                         src     dst            ntiles ks  N    K   rowS kS b0 dS
    pack_b_frag<<<136, 32, 0, stream>>>(stft_w, wp + OFF_STFT_H, 17, 8, 258, 256, 256, 1, 0, 0);
    pack_b_frag<<<120, 32, 0, stream>>>(e1_w,   wp + OFF_E1_H,    8, 5, 128, 129, 387, 3, 0, 1);
    pack_b_frag<<< 48, 32, 0, stream>>>(e2_w,   wp + OFF_E2_H,    4, 4,  64, 128, 384, 3, 0, 1);
    pack_b_frag<<< 16, 32, 0, stream>>>(e3_w,   wp + OFF_E3_H,    4, 2,  64,  64, 192, 3, 1, 1);
    pack_b_frag<<< 16, 32, 0, stream>>>(e4_w,   wp + OFF_E4_H,    8, 2, 128,  64, 192, 3, 1, 0);
    pack_b_frag<<<128, 32, 0, stream>>>(w_ih,   wp + OFF_LSTM_H, 32, 4, 512, 128, 128, 1, 0, 0);

    // -------- fused forward pass: 16 batch rows per workgroup ----------------
    (void)hipFuncSetAttribute((const void*)vad_fused,
                              hipFuncAttributeMaxDynamicSharedMemorySize,
                              (int)SMEM_BYTES);
    int nrows   = in_sizes[0] / 512;   // 65536
    int nblocks = nrows / 16;          // 4096
    vad_fused<<<nblocks, 128, SMEM_BYTES, stream>>>(
        data, wp, e1_b, e2_b, e3_b, e4_b, b_ih, b_hh, dec_w, dec_b, out);
}
